// MemModulePlastic_18811956757049
// MI455X (gfx1250) — matrix operand order repaired and checked by reading, not compiled, not run on hardware
//
#include <hip/hip_runtime.h>
#include <hip/hip_fp16.h>

// MemModulePlastic: s = state@P ; argmax_h(s . memories[h]) ; out = logits[argmax]
// BATCH=2048, FLAT_IN=512, PROJ=256, HEADS=16384, ACT=64

typedef __attribute__((ext_vector_type(16))) _Float16 v16h;
typedef __attribute__((ext_vector_type(8)))  _Float16 v8h;
typedef __attribute__((ext_vector_type(4)))  _Float16 v4h;
typedef __attribute__((ext_vector_type(8)))  float    v8f;

#define BATCH    2048
#define FLAT_IN  512
#define PROJ     256
#define HEADS    16384
#define ACT      64
#define NH_CHUNKS 16
#define CHUNK_HEADS (HEADS / NH_CHUNKS)   // 1024 heads per block column
#define ITER_HEADS 128                    // heads staged in LDS per iteration
#define NITER (CHUNK_HEADS / ITER_HEADS)  // 8
#define B_PITCH 264                       // halves; 528B pitch: 16B-aligned rows,
                                          // 4-bank stride -> conflict-free b128 reads

static __device__ __forceinline__ v16h cat16(v8h a, v8h b) {
  return __builtin_shufflevector(a, b, 0, 1, 2, 3, 4, 5, 6, 7,
                                 8, 9, 10, 11, 12, 13, 14, 15);
}

// -------------------------------------------------------------------------
// Kernel 1: s16[2048x256] (f16) = state[2048x512] @ P[512x256]
// One wave computes one 16x16 tile, K-loop of 512 in steps of 32 (16 WMMAs).
// A fragment = two contiguous float4-pairs per lane (ISA 16-bit A layout).
// -------------------------------------------------------------------------
__global__ __launch_bounds__(256) void proj_kernel(
    const float* __restrict__ state, const float* __restrict__ P,
    _Float16* __restrict__ s16) {
  const int lane = threadIdx.x & 31;
  const int wave = threadIdx.x >> 5;
  const int tile = blockIdx.x * 8 + wave;      // 2048 tiles total
  const int tm = tile >> 4;                    // 0..127  (M tile)
  const int tn = tile & 15;                    // 0..15   (N tile)
  const int l15 = lane & 15;
  const bool hi = lane >= 16;

  const int arow = tm * 16 + l15;              // A row for this lane
  const int bcol = tn * 16 + l15;              // B column for this lane

  v8f c = {};
  for (int k0 = 0; k0 < FLAT_IN; k0 += 32) {
    // ---- A fragment: lanes<16 hold K = k0+{0..7, 16..23}; hi lanes +8
    const float4* ap =
        reinterpret_cast<const float4*>(state + (size_t)arow * FLAT_IN + k0 + (hi ? 8 : 0));
    const float4 a0 = ap[0];   // K+0..3
    const float4 a1 = ap[1];   // K+4..7
    const float4 a2 = ap[4];   // K+16..19
    const float4 a3 = ap[5];   // K+20..23
    v16h a;
    a[0]  = (_Float16)a0.x; a[1]  = (_Float16)a0.y;
    a[2]  = (_Float16)a0.z; a[3]  = (_Float16)a0.w;
    a[4]  = (_Float16)a1.x; a[5]  = (_Float16)a1.y;
    a[6]  = (_Float16)a1.z; a[7]  = (_Float16)a1.w;
    a[8]  = (_Float16)a2.x; a[9]  = (_Float16)a2.y;
    a[10] = (_Float16)a2.z; a[11] = (_Float16)a2.w;
    a[12] = (_Float16)a3.x; a[13] = (_Float16)a3.y;
    a[14] = (_Float16)a3.z; a[15] = (_Float16)a3.w;
    // ---- B fragment: lanes<16 K=k0+{0..7, 16..23} ; hi lanes +8
    v16h b;
    const int kb = k0 + (hi ? 8 : 0);
#pragma unroll
    for (int j = 0; j < 8; ++j) {
      const int kk = kb + 2 * j + (j >= 4 ? 8 : 0);
      b[2 * j]     = (_Float16)P[(size_t)kk * PROJ + bcol];
      b[2 * j + 1] = (_Float16)P[(size_t)(kk + 1) * PROJ + bcol];
    }
    c = __builtin_amdgcn_wmma_f32_16x16x32_f16(false, a, false, b,
                                               (short)0, c, false, false);
  }
  // C layout: VGPR j = row (j | j+8), col = lane&15
#pragma unroll
  for (int j = 0; j < 8; ++j) {
    const int row = tm * 16 + (hi ? (8 + j) : j);
    s16[(size_t)row * PROJ + bcol] = (_Float16)c[j];
  }
}

// -------------------------------------------------------------------------
// Kernel 2: fused similarities + partial argmax.
// Block = 64 rows (blockIdx.x) x 1024 heads (blockIdx.y), 8 waves:
//   wave = m_sub (4 subtiles of 16 rows) x h_sub (2 head halves of 64).
// memories staged per-iteration into LDS (128 heads x 256 K, f32->f16).
// B-tile in LDS is shared by the 4 m_sub waves -> 4x L2 traffic reduction.
// -------------------------------------------------------------------------
__global__ __launch_bounds__(256) void sim_argmax_kernel(
    const _Float16* __restrict__ s16, const float* __restrict__ memories,
    float* __restrict__ pmax, int* __restrict__ pidx) {
  __shared__ _Float16 ldsB[ITER_HEADS * B_PITCH]; // 128*264*2 = 66 KB
  __shared__ float wmax[64 * 2];
  __shared__ int   widx[64 * 2];

  const int tid  = threadIdx.x;
  const int lane = tid & 31;
  const int wave = tid >> 5;
  const int l15  = lane & 15;
  const bool hi  = lane >= 16;
  const int m_sub = wave & 3;
  const int h_sub = wave >> 2;

  const int rowBase = blockIdx.x * 64 + m_sub * 16;
  const int H0 = blockIdx.y * CHUNK_HEADS;

  // ---- A fragments for all 8 K-steps: 16 x global_load_b128 (64 VGPRs)
  v16h afrag[8];
  {
    const _Float16* srow = s16 + (size_t)(rowBase + l15) * PROJ;
#pragma unroll
    for (int t = 0; t < 8; ++t) {
      const int k0 = t * 32 + (hi ? 8 : 0);
      const v8h a0 = *reinterpret_cast<const v8h*>(srow + k0);       // K+0..7
      const v8h a1 = *reinterpret_cast<const v8h*>(srow + k0 + 16);  // K+16..23
      afrag[t] = cat16(a0, a1);
    }
  }

  float mx[8];
  int   ix[8];
#pragma unroll
  for (int j = 0; j < 8; ++j) { mx[j] = -3.402823e38f; ix[j] = 0; }

  const float4* membase = reinterpret_cast<const float4*>(memories);

  for (int it = 0; it < NITER; ++it) {
    __syncthreads();   // protect ldsB from previous iteration's readers
    // ---- cooperative stage: 128 heads x 256 K f32 -> f16 LDS (ds_store_b64)
    const float4* msrc = membase + (size_t)(H0 + it * ITER_HEADS) * (PROJ / 4);
#pragma unroll 8
    for (int p = 0; p < 32; ++p) {
      const int idx4 = tid + p * 256;      // 0..8191 float4s
      const int hl = idx4 >> 6;            // local head (64 float4 per row)
      const int kk = (idx4 & 63) * 4;      // K offset
      const float4 v = msrc[idx4];
      v4h pk;
      pk.x = (_Float16)v.x; pk.y = (_Float16)v.y;
      pk.z = (_Float16)v.z; pk.w = (_Float16)v.w;
      *reinterpret_cast<v4h*>(&ldsB[hl * B_PITCH + kk]) = pk;
    }
    // prefetch next iteration's chunk toward L2/L0
    if (it + 1 < NITER) {
      __builtin_prefetch(
          (const void*)(memories + (size_t)(H0 + (it + 1) * ITER_HEADS) * PROJ + tid * 128),
          0, 1);
    }
    __syncthreads();

    // ---- each wave: four 16x16 output tiles (its 64-head half)
#pragma unroll
    for (int nt = 0; nt < 4; ++nt) {
      const int hloc = h_sub * 64 + nt * 16 + l15;     // local head for lane
      const int hg = H0 + it * ITER_HEADS + hloc;      // global head index
      const _Float16* brow = &ldsB[hloc * B_PITCH];
      v8f c = {};
#pragma unroll
      for (int t = 0; t < 8; ++t) {
        const int kb = t * 32 + (hi ? 8 : 0);
        const v8h b0 = *reinterpret_cast<const v8h*>(brow + kb);      // ds_load_b128
        const v8h b1 = *reinterpret_cast<const v8h*>(brow + kb + 16); // ds_load_b128
        c = __builtin_amdgcn_wmma_f32_16x16x32_f16(false, afrag[t], false,
                                                   cat16(b0, b1),
                                                   (short)0, c, false, false);
      }
      // running per-row argmax (this lane's head column)
#pragma unroll
      for (int j = 0; j < 8; ++j) {
        if (c[j] > mx[j]) { mx[j] = c[j]; ix[j] = hg; }
      }
    }
  }

  // ---- cross-lane argmax within each 16-lane group (head dimension)
#pragma unroll
  for (int m = 1; m <= 8; m <<= 1) {
#pragma unroll
    for (int j = 0; j < 8; ++j) {
      const float omx = __shfl_xor(mx[j], m, 32);
      const int   oix = __shfl_xor(ix[j], m, 32);
      if (omx > mx[j] || (omx == mx[j] && oix < ix[j])) { mx[j] = omx; ix[j] = oix; }
    }
  }
  // lanes 0 and 16 hold reduced results for rows j and j+8 respectively
  if (l15 == 0) {
#pragma unroll
    for (int j = 0; j < 8; ++j) {
      const int rloc = m_sub * 16 + (hi ? (8 + j) : j);
      wmax[rloc * 2 + h_sub] = mx[j];
      widx[rloc * 2 + h_sub] = ix[j];
    }
  }
  __syncthreads();
  // combine the two head-halves, emit partial per (row, H-chunk)
  if (tid < 64) {
    const float m0 = wmax[tid * 2 + 0]; const int i0 = widx[tid * 2 + 0];
    const float m1 = wmax[tid * 2 + 1]; const int i1 = widx[tid * 2 + 1];
    const bool take1 = (m1 > m0) || (m1 == m0 && i1 < i0);
    const int grow = blockIdx.x * 64 + tid;
    pmax[(size_t)grow * NH_CHUNKS + blockIdx.y] = take1 ? m1 : m0;
    pidx[(size_t)grow * NH_CHUNKS + blockIdx.y] = take1 ? i1 : i0;
  }
}

// -------------------------------------------------------------------------
// Kernel 3: reduce 16 partials per row, gather logits[closest] -> out.
// 4 threads per row; each copies 16 floats (4x float4).
// -------------------------------------------------------------------------
__global__ __launch_bounds__(256) void select_gather_kernel(
    const float* __restrict__ pmax, const int* __restrict__ pidx,
    const float* __restrict__ logits, float* __restrict__ out) {
  const int tid = threadIdx.x;
  const int row = blockIdx.x * 64 + (tid >> 2);
  const int q   = tid & 3;
  float best = -3.402823e38f;
  int bidx = 0;
#pragma unroll
  for (int c = 0; c < NH_CHUNKS; ++c) {
    const float m = pmax[(size_t)row * NH_CHUNKS + c];
    const int   i = pidx[(size_t)row * NH_CHUNKS + c];
    if (m > best || (m == best && i < bidx)) { best = m; bidx = i; }
  }
  const float4* src = reinterpret_cast<const float4*>(logits + (size_t)bidx * ACT) + q * 4;
  float4* dst       = reinterpret_cast<float4*>(out + (size_t)row * ACT) + q * 4;
#pragma unroll
  for (int p = 0; p < 4; ++p) dst[p] = src[p];
}

// -------------------------------------------------------------------------
extern "C" void kernel_launch(void* const* d_in, const int* in_sizes, int n_in,
                              void* d_out, int out_size, void* d_ws, size_t ws_size,
                              hipStream_t stream) {
  (void)in_sizes; (void)n_in; (void)out_size; (void)ws_size;
  const float* state    = (const float*)d_in[0];
  const float* P        = (const float*)d_in[1];
  const float* memories = (const float*)d_in[2];
  const float* logits   = (const float*)d_in[3];
  float* out = (float*)d_out;

  char* ws = (char*)d_ws;
  _Float16* s16 = (_Float16*)ws;                                   // 1 MB
  float* pmax   = (float*)(ws + (size_t)BATCH * PROJ * 2);         // 128 KB
  int*   pidx   = (int*)(ws + (size_t)BATCH * PROJ * 2
                            + (size_t)BATCH * NH_CHUNKS * 4);      // 128 KB

  proj_kernel<<<256, 256, 0, stream>>>(state, P, s16);
  dim3 g2(BATCH / 64, NH_CHUNKS);
  sim_argmax_kernel<<<g2, 256, 0, stream>>>(s16, memories, pmax, pidx);
  select_gather_kernel<<<BATCH / 64, 256, 0, stream>>>(pmax, pidx, logits, out);
}
